// SQMF_26499948216761
// MI455X (gfx1250) — matrix operand order repaired and checked by reading, not compiled, not run on hardware
//
#include <hip/hip_runtime.h>
#include <hip/hip_bf16.h>

// ---------------------------------------------------------------------------
// SQMF fused kernel for MI455X (gfx1250, wave32, WMMA).
//   psi  = phi[ii]*phi[jj]           (2080, n)  -- generated on the fly in LDS
//   quad = theta^T @ psi             (32, n)    -- WMMA f32_16x16x32_f16
//   z    = [phi; quad]               (96, n)    -- kept transposed in LDS
//   out  = c + Q @ z                 (2048, n)  -- WMMA f32_16x16x32_f16
// Roofline: 512MB out + 16MB phi ~ 23us floor @ 23.3 TB/s; ~35 GFLOP in f16
// WMMA is far below that floor. Q (768KB) / theta (266KB) are L2-resident.
// Phase B double-buffers Q panels L2->LDS with global_load_async_to_lds_b128
// (ASYNCcnt path), overlapping the copy with the WMMA stream.
// ---------------------------------------------------------------------------

typedef _Float16 f16;
typedef f16   v16h __attribute__((ext_vector_type(16)));
typedef float v8f  __attribute__((ext_vector_type(8)));

#define Dd   2048
#define dd   64
#define Ss   32
#define Nn   65536
#define NB   128          // columns per workgroup
#define TRI  2080         // dd*(dd+1)/2, = 65 * 32 exactly

union FragU { v16h v; uint4 q[2]; };

// A-matrix 16x32 f16 fragment (M=lane%16, K-octets selected by half):
//   elems 0..7  -> K = kb + half*8 + (0..7)
//   elems 8..15 -> K = kb + 16 + half*8 + (0..7)
__device__ __forceinline__ v16h loadA(const f16* row, int kb, int half) {
  FragU u;
  u.q[0] = *(const uint4*)(row + kb + half * 8);
  u.q[1] = *(const uint4*)(row + kb + 16 + half * 8);
  return u.v;
}

// Same A fragment but from an f32 LDS row: 4x ds_load_b128 + 8x v_cvt_pk.
__device__ __forceinline__ v16h loadA32(const float* row, int kb, int half) {
  const float* p0 = row + kb + half * 8;
  const float* p1 = row + kb + 16 + half * 8;
  v16h r;
  #pragma unroll
  for (int e = 0; e < 8; ++e) r[e] = (f16)p0[e];
  #pragma unroll
  for (int e = 0; e < 8; ++e) r[8 + e] = (f16)p1[e];
  return r;
}

// B-matrix 32x16 f16 fragment from a [N][K]-contiguous LDS row:
//   lane n = lane%16, elems 0..7 -> K = kb + half*8 + e, elems 8..15 -> K = kb + 16 + half*8 + (e-8)
__device__ __forceinline__ v16h loadB(const f16* row, int kb, int half) {
  FragU u;
  u.q[0] = *(const uint4*)(row + kb + half * 8);
  u.q[1] = *(const uint4*)(row + kb + 16 + half * 8);
  return u.v;
}

__device__ __forceinline__ v8f wmma16(v16h a, v16h b, v8f c) {
  return __builtin_amdgcn_wmma_f32_16x16x32_f16(
      /*neg_a=*/false, a, /*neg_b=*/false, b,
      /*c_mod=*/(short)0, c, /*reuse_a=*/false, /*reuse_b=*/false);
}

// Async L2->LDS copy of one 16B chunk per lane (ASYNCcnt-tracked).
__device__ __forceinline__ void asyncCopy16(unsigned int lds_byte_addr,
                                            const void* gaddr) {
  asm volatile("global_load_async_to_lds_b128 %0, %1, off"
               :: "v"(lds_byte_addr), "v"(gaddr) : "memory");
}

// Issue one 64x96 f32 Q panel (24KB = 1536 x 16B): 6 async ops per thread.
__device__ __forceinline__ void issueQPanel(const float4* Q4, int m0,
                                            float* dstLds, int tid) {
  unsigned int base = (unsigned int)(uintptr_t)dstLds;  // low 32b = LDS offset
  #pragma unroll
  for (int q2 = 0; q2 < 6; ++q2) {
    int idx = tid + 256 * q2;            // 0..1535 float4 chunks
    int mm = idx / 24;
    int kq = idx - mm * 24;
    asyncCopy16(base + (unsigned int)idx * 16u,
                (const void*)(Q4 + (size_t)(m0 + mm) * 24 + kq));
  }
}

__global__ __launch_bounds__(256) void sqmf_fused_kernel(
    const float* __restrict__ cvec,   // (2048,1)
    const float* __restrict__ Q,      // (2048,96) row-major
    const float* __restrict__ phi,    // (64,65536) row-major
    const float* __restrict__ theta,  // (2080,32) row-major
    float* __restrict__ out)          // (2048,65536) row-major
{
  __shared__ __align__(16) f16   zT[NB][96];    // [col][k]: k<64 phi, k>=64 quad (24KB)
  __shared__ __align__(16) f16   psiT[NB][32];  // [col][k] psi K-tile            (8KB)
  __shared__ __align__(16) f16   thT[32][32];   // [s][k] theta^T K-tile          (2KB)
  __shared__ __align__(16) float Qf[2][64][96]; // double-buffered f32 Q panels   (48KB)
  __shared__ float cP[64];
  __shared__ unsigned char iiL[TRI], jjL[TRI];  // triangular index LUT (4KB)

  const int tid  = threadIdx.x;
  const int lane = tid & 31;
  const int wave = tid >> 5;        // 8 waves
  const int half = lane >> 4;       // 0/1
  const int lm   = lane & 15;
  const int c0   = blockIdx.x * NB;

  const float4* Q4 = (const float4*)Q;          // Q row = 24 float4

  // Kick off the first Q panel now: it streams L2->LDS under all of Phase A.
  issueQPanel(Q4, 0, &Qf[0][0][0], tid);

  // ---- build upper-triangular (i,j) LUT: r -> (i, j), i<=j, row-major ----
  for (int r = tid; r < TRI; r += 256) {
    int i = 0, start = 0;
    while (start + (dd - i) <= r) { start += dd - i; ++i; }
    iiL[r] = (unsigned char)i;
    jjL[r] = (unsigned char)(i + (r - start));
  }

  // ---- stage phi block transposed into zT[col][0..63] (f32 -> f16) ----
  const float4* phi4 = (const float4*)phi;      // row = Nn/4 float4
  #pragma unroll
  for (int it = 0; it < 8; ++it) {
    int idx = tid + 256 * it;                   // 0..2047
    int row = idx >> 5;                         // k: 0..63
    int cq  = idx & 31;                         // float4 within 128 cols
    float4 v = phi4[(size_t)row * (Nn / 4) + (c0 >> 2) + cq];
    int cc = cq * 4;
    zT[cc + 0][row] = (f16)v.x;
    zT[cc + 1][row] = (f16)v.y;
    zT[cc + 2][row] = (f16)v.z;
    zT[cc + 3][row] = (f16)v.w;
  }
  __syncthreads();

  // =========================== Phase A: quad ============================
  // quad(32 x 128) = theta^T(32 x 2080) @ psi(2080 x 128), K-tiled by 32.
  // Wave w: M-tile = w>>2 (s rows), N-tiles nt0, nt0+1.
  const int mtileA = wave >> 2;
  const int nt0    = (wave & 3) * 2;
  v8f acc0 = {};
  v8f acc1 = {};

  for (int ks = 0; ks < TRI / 32; ++ks) {       // 65 steps, no remainder
    const int rbase = ks * 32;

    // psi tile: 128 cols x 32 k, 16 values per thread, contiguous stores
    {
      int col = tid >> 1;
      int seg = (tid & 1) * 16;
      const f16* zr = &zT[col][0];
      #pragma unroll
      for (int e = 0; e < 16; ++e) {
        int r = rbase + seg + e;
        float pi = (float)zr[iiL[r]];
        float pj = (float)zr[jjL[r]];
        psiT[col][seg + e] = (f16)(pi * pj);
      }
    }
    // theta^T tile: 32 s x 32 k (coalesced f32 reads along s)
    #pragma unroll
    for (int q2 = 0; q2 < 4; ++q2) {
      int idx = tid + 256 * q2;                 // 0..1023
      int s = idx & 31;
      int k = idx >> 5;
      thT[s][k] = (f16)theta[(size_t)(rbase + k) * 32 + s];
    }
    __syncthreads();

    v16h a  = loadA(&thT[mtileA * 16 + lm][0], 0, half);
    v16h b0 = loadB(&psiT[nt0 * 16 + lm][0], 0, half);
    v16h b1 = loadB(&psiT[(nt0 + 1) * 16 + lm][0], 0, half);
    acc0 = wmma16(a, b0, acc0);
    acc1 = wmma16(a, b1, acc1);
    __syncthreads();
  }

  // write quad into zT[col][64..95] (C layout: m = 8*half + r, n = lm)
  #pragma unroll
  for (int r = 0; r < 8; ++r) {
    int srow = mtileA * 16 + half * 8 + r;      // 0..31
    zT[nt0 * 16 + lm][64 + srow]       = (f16)acc0[r];
    zT[(nt0 + 1) * 16 + lm][64 + srow] = (f16)acc1[r];
  }
  __syncthreads();

  // =========================== Phase B: out =============================
  // out(2048 x 128) = c + Q(2048 x 96) @ z(96 x 128).
  // Wave w owns N-tile w; its 3 z B-fragments are loop-invariant -> hoist.
  // Q panels arrive via double-buffered async L2->LDS copies.
  const int ntile = wave;
  const f16* zrow = &zT[ntile * 16 + lm][0];
  v16h zb0 = loadB(zrow, 0, half);
  v16h zb1 = loadB(zrow, 32, half);
  v16h zb2 = loadB(zrow, 64, half);

  for (int ms = 0; ms < Dd / 64; ++ms) {        // 32 M-panels of 64 rows
    const int m0 = ms * 64;

    if (ms + 1 < Dd / 64) {
      // Stream next panel while this one is consumed; <=6 outstanding means
      // the previous panel's 6 in-order async ops have all landed.
      issueQPanel(Q4, m0 + 64, &Qf[(ms + 1) & 1][0][0], tid);
      asm volatile("s_wait_asynccnt 0x6" ::: "memory");
    } else {
      asm volatile("s_wait_asynccnt 0x0" ::: "memory");
    }
    if (tid < 64) cP[tid] = cvec[m0 + tid];
    __syncthreads();   // panel ms visible to all waves

    const float (*Qcur)[96] = Qf[ms & 1];
    #pragma unroll
    for (int mt = 0; mt < 4; ++mt) {
      const float* qrow = &Qcur[mt * 16 + lm][0];
      v8f acc = {};
      acc = wmma16(loadA32(qrow, 0, half),  zb0, acc);
      acc = wmma16(loadA32(qrow, 32, half), zb1, acc);
      acc = wmma16(loadA32(qrow, 64, half), zb2, acc);

      const int gm = m0 + mt * 16 + half * 8;
      const int gc = c0 + ntile * 16 + lm;
      #pragma unroll
      for (int r = 0; r < 8; ++r) {
        out[(size_t)(gm + r) * Nn + gc] = cP[mt * 16 + half * 8 + r] + acc[r];
      }
    }
    __syncthreads();   // all reads of Qf[ms&1] done before it is re-filled
  }
}

extern "C" void kernel_launch(void* const* d_in, const int* in_sizes, int n_in,
                              void* d_out, int out_size, void* d_ws, size_t ws_size,
                              hipStream_t stream) {
  const float* c     = (const float*)d_in[0];   // (2048,1)
  const float* Q     = (const float*)d_in[1];   // (2048,96)
  const float* phi   = (const float*)d_in[2];   // (64,65536)
  const float* theta = (const float*)d_in[3];   // (2080,32)
  float* out = (float*)d_out;                   // (2048,65536)

  dim3 grid(Nn / NB);                           // 512 workgroups
  dim3 block(256);                              // 8 wave32 waves
  sqmf_fused_kernel<<<grid, block, 0, stream>>>(c, Q, phi, theta, out);
}
